// HeteroGNN_37726992728705
// MI455X (gfx1250) — matrix operand order repaired and checked by reading, not compiled, not run on hardware
//
#include <hip/hip_runtime.h>
#include <hip/hip_bf16.h>

// ---------------------------------------------------------------------------
// HeteroGNN (SAGEConv mean-aggr, 2 layers, 8 edge types) for MI455X / gfx1250
// GEMMs run on v_wmma_f32_16x16x32_bf16 (f32 accumulate). Scatter phase uses
// plain float atomics (L2-resident on a 192MB L2).
// ---------------------------------------------------------------------------

typedef __attribute__((ext_vector_type(16))) __bf16 v16bf;
typedef __attribute__((ext_vector_type(8)))  __bf16 v8bf;
typedef __attribute__((ext_vector_type(8)))  float  v8f;

#define NN     32768
#define HH     128
#define NEDGE  400000
#define ETY    8
#define NLAY   2
#define OUTC   64

__device__ __forceinline__ v16bf cat8(v8bf a, v8bf b) {
  return __builtin_shufflevector(a, b, 0,1,2,3,4,5,6,7,8,9,10,11,12,13,14,15);
}

// ---------------- utility kernels ----------------

__global__ void zero_kernel(float* __restrict__ p, int n) {
  int i = blockIdx.x * blockDim.x + threadIdx.x;
  if (i < n) p[i] = 0.f;
}

// x[row][:] = emb[idx[row]][:]   (128 floats per row, 4 per thread)
__global__ void gather_embed(const float* __restrict__ emb,
                             const int* __restrict__ idx,
                             float* __restrict__ x) {
  int t   = blockIdx.x * blockDim.x + threadIdx.x;   // N*32 threads
  int row = t >> 5;
  int q   = (t & 31) * 4;
  float4 v = *(const float4*)(emb + (long)idx[row] * HH + q);
  *(float4*)(x + (long)row * HH + q) = v;
}

// edge scatter: m[di] += x[si], deg[di] += 1
__global__ void scatter_add(const int* __restrict__ si, const int* __restrict__ di,
                            const float* __restrict__ xsrc,
                            float* __restrict__ m, float* __restrict__ deg, int ne) {
  int t = blockIdx.x * blockDim.x + threadIdx.x;
  int j = t >> 5;
  if (j >= ne) return;
  int lane = t & 31;
  int s = si[j], d = di[j];
  if (lane == 0) atomicAdd(&deg[d], 1.0f);
  float4 v = *(const float4*)(xsrc + (long)s * HH + lane * 4);
  float* mp = m + (long)d * HH + lane * 4;
  atomicAdd(mp + 0, v.x); atomicAdd(mp + 1, v.y);
  atomicAdd(mp + 2, v.z); atomicAdd(mp + 3, v.w);
}

__global__ void mean_div(float* __restrict__ m, const float* __restrict__ deg) {
  int t = blockIdx.x * blockDim.x + threadIdx.x;   // N*H threads
  m[t] /= fmaxf(deg[t >> 7], 1.0f);
}

__global__ void relu_kernel(const float* __restrict__ acc, float* __restrict__ x, int n) {
  int i = blockIdx.x * blockDim.x + threadIdx.x;
  if (i < n) x[i] = fmaxf(acc[i], 0.f);
}

// transpose-convert: nmat matrices of (K x Nc) f32 row-major ->
// bf16 column-major: o[mat][n*K + k] = w[mat][k*Nc + n]
__global__ void wconv_t(const float* __restrict__ w, __bf16* __restrict__ o,
                        int K, int Nc, long tot) {
  long t = (long)blockIdx.x * blockDim.x + threadIdx.x;
  if (t >= tot) return;
  long mk = (long)K * Nc;
  int mat = (int)(t / mk);
  int r   = (int)(t % mk);
  int n   = r / K;
  int k   = r % K;
  o[t] = (__bf16)w[(long)mat * mk + (long)k * Nc + n];
}

// ---------------- fused SAGEConv GEMM kernel ----------------
// Block: 16 dst rows, 256 threads = 8 waves, wave w owns cols [16w,16w+16).
// o(16x128) = m_blk @ Wl + x_blk @ Wr + bl ; acc += o / max(||o||_row, 1e-12)
// WlT / WrT are bf16, column-major: WT[n*128 + k].
__global__ __launch_bounds__(256)
void sage_gemm_norm_acc(const float* __restrict__ m,
                        const float* __restrict__ xdst,
                        const __bf16* __restrict__ WlT,
                        const __bf16* __restrict__ WrT,
                        const float* __restrict__ bl,
                        float* __restrict__ acc) {
  __shared__ __bf16 lm[16 * HH];
  __shared__ __bf16 lx[16 * HH];
  __shared__ float  lo[16 * HH];
  __shared__ float  lnorm[16];

  const int tid  = threadIdx.x;
  const int row0 = blockIdx.x * 16;

  if (tid < 16) lnorm[tid] = 0.f;
  for (int i = tid; i < 16 * HH; i += 256) {
    int r = i >> 7, c = i & 127;
    long g = (long)(row0 + r) * HH + c;
    lm[i] = (__bf16)m[g];
    lx[i] = (__bf16)xdst[g];
  }
  __syncthreads();

  const int wave = tid >> 5;
  const int lane = tid & 31;
  const int half = lane >> 4;
  const int nloc = lane & 15;
  const int ncol = wave * 16 + nloc;

  const __bf16* blcol = WlT + (long)ncol * HH;
  const __bf16* brcol = WrT + (long)ncol * HH;

  v8f c = {};
  #pragma unroll
  for (int kc = 0; kc < 4; ++kc) {
    // A fragments (16-bit A layout: lanes 0-15 K 0-7/16-23, lanes 16-31 K 8-15/24-31)
    const __bf16* am_p = &lm[(lane & 15) * HH + kc * 32 + half * 8];
    const __bf16* ax_p = &lx[(lane & 15) * HH + kc * 32 + half * 8];
    v16bf am = cat8(*(const v8bf*)(am_p), *(const v8bf*)(am_p + 16));
    v16bf ax = cat8(*(const v8bf*)(ax_p), *(const v8bf*)(ax_p + 16));
    // B fragments (lanes 0-15: K 0-7/16-23, lanes 16-31: K 8-15/24-31) — same order as A
    const __bf16* bl_p = blcol + kc * 32 + half * 8;
    const __bf16* br_p = brcol + kc * 32 + half * 8;
    v16bf bwl = cat8(*(const v8bf*)(bl_p), *(const v8bf*)(bl_p + 16));
    v16bf bwr = cat8(*(const v8bf*)(br_p), *(const v8bf*)(br_p + 16));
    c = __builtin_amdgcn_wmma_f32_16x16x32_bf16(false, am, false, bwl, (short)0, c, false, false);
    c = __builtin_amdgcn_wmma_f32_16x16x32_bf16(false, ax, false, bwr, (short)0, c, false, false);
  }

  // bias + stage tile to LDS (C/D layout: VGPR r -> row half*8+r, col ncol)
  float bias = bl[ncol];
  #pragma unroll
  for (int r = 0; r < 8; ++r)
    lo[(half * 8 + r) * HH + ncol] = c[r] + bias;
  __syncthreads();

  // row L2 norms: 16 partials per row
  {
    int r = tid >> 4, off = (tid & 15) * 8;
    float s = 0.f;
    #pragma unroll
    for (int i = 0; i < 8; ++i) { float v = lo[r * HH + off + i]; s += v * v; }
    atomicAdd(&lnorm[r], s);
  }
  __syncthreads();
  if (tid < 16) lnorm[tid] = fmaxf(sqrtf(lnorm[tid]), 1e-12f);
  __syncthreads();

  for (int i = tid; i < 16 * HH; i += 256) {
    int r = i >> 7;
    acc[(long)(row0 + r) * HH + (i & 127)] += lo[i] / lnorm[r];
  }
}

// ---------------- final projection: out = x_reaction @ W_out + b_out ------
// Block: 16 rows, 128 threads = 4 waves covering 64 output cols.
__global__ __launch_bounds__(128)
void out_gemm(const float* __restrict__ xr, const __bf16* __restrict__ WoT,
              const float* __restrict__ bo, float* __restrict__ out) {
  __shared__ __bf16 lx[16 * HH];
  const int tid  = threadIdx.x;
  const int row0 = blockIdx.x * 16;

  for (int i = tid; i < 16 * HH; i += 128)
    lx[i] = (__bf16)xr[(long)(row0 + (i >> 7)) * HH + (i & 127)];
  __syncthreads();

  const int wave = tid >> 5;
  const int lane = tid & 31;
  const int half = lane >> 4;
  const int nloc = lane & 15;
  const int ncol = wave * 16 + nloc;          // 0..63
  const __bf16* bcol = WoT + (long)ncol * HH;

  v8f c = {};
  #pragma unroll
  for (int kc = 0; kc < 4; ++kc) {
    const __bf16* ap = &lx[(lane & 15) * HH + kc * 32 + half * 8];
    v16bf a = cat8(*(const v8bf*)(ap), *(const v8bf*)(ap + 16));
    const __bf16* bp = bcol + kc * 32 + half * 8;
    v16bf b = cat8(*(const v8bf*)(bp), *(const v8bf*)(bp + 16));
    c = __builtin_amdgcn_wmma_f32_16x16x32_bf16(false, a, false, b, (short)0, c, false, false);
  }
  float bias = bo[ncol];
  #pragma unroll
  for (int r = 0; r < 8; ++r)
    out[(long)(row0 + half * 8 + r) * OUTC + ncol] = c[r] + bias;
}

// ---------------- host-side orchestration ----------------

extern "C" void kernel_launch(void* const* d_in, const int* in_sizes, int n_in,
                              void* d_out, int out_size, void* d_ws, size_t ws_size,
                              hipStream_t stream) {
  const int*   idx[4]  = { (const int*)d_in[0], (const int*)d_in[1],
                           (const int*)d_in[2], (const int*)d_in[3] };
  const int*   eidx    = (const int*)d_in[4];
  const float* emb[4]  = { (const float*)d_in[5], (const float*)d_in[6],
                           (const float*)d_in[7], (const float*)d_in[8] };
  const float* Wl      = (const float*)d_in[9];
  const float* bl      = (const float*)d_in[10];
  const float* Wr      = (const float*)d_in[11];
  const float* Wo      = (const float*)d_in[12];
  const float* bo      = (const float*)d_in[13];
  float*       out     = (float*)d_out;

  // workspace layout (floats)
  const long NH = (long)NN * HH;
  float* ws   = (float*)d_ws;
  float* x0   = ws;                 // 4 x (N*H): current features per type
  float* acc0 = ws + 4 * NH;        // 4 x (N*H): accumulators per type
  float* m    = ws + 8 * NH;        // N*H  : segment mean scratch
  float* deg  = ws + 9 * NH;        // N    : degree counts
  __bf16* wlT = (__bf16*)(deg + NN);                     // L*E*H*H bf16 (transposed)
  __bf16* wrT = wlT + (long)NLAY * ETY * HH * HH;
  __bf16* woT = wrT + (long)NLAY * ETY * HH * HH;        // OUTC*H bf16

  static const int SRC[ETY] = {0, 1, 3, 0, 1, 2, 2, 3};
  static const int DST[ETY] = {2, 2, 2, 3, 3, 0, 1, 0};

  // 1) weight transpose-convert to bf16 column-major
  {
    long tot = (long)NLAY * ETY * HH * HH;
    int blocks = (int)((tot + 255) / 256);
    wconv_t<<<blocks, 256, 0, stream>>>(Wl, wlT, HH, HH, tot);
    wconv_t<<<blocks, 256, 0, stream>>>(Wr, wrT, HH, HH, tot);
    long tot2 = (long)HH * OUTC;
    wconv_t<<<(int)((tot2 + 255) / 256), 256, 0, stream>>>(Wo, woT, HH, OUTC, tot2);
  }

  // 2) embedding gather (reaction/complex index is all-zeros -> row 0)
  for (int t = 0; t < 4; ++t)
    gather_embed<<<NN * 32 / 256, 256, 0, stream>>>(emb[t], idx[t], x0 + t * NH);

  // 3) layers
  for (int l = 0; l < NLAY; ++l) {
    zero_kernel<<<(int)(4 * NH / 256), 256, 0, stream>>>(acc0, (int)(4 * NH));
    for (int e = 0; e < ETY; ++e) {
      const int* si = eidx + ((long)e * 2 + 0) * NEDGE;
      const int* di = eidx + ((long)e * 2 + 1) * NEDGE;
      // m and deg are contiguous -> one zero pass
      zero_kernel<<<(int)((NH + NN + 255) / 256), 256, 0, stream>>>(m, (int)(NH + NN));
      scatter_add<<<(NEDGE * 32 + 255) / 256, 256, 0, stream>>>(
          si, di, x0 + SRC[e] * NH, m, deg, NEDGE);
      mean_div<<<(int)(NH / 256), 256, 0, stream>>>(m, deg);
      sage_gemm_norm_acc<<<NN / 16, 256, 0, stream>>>(
          m, x0 + DST[e] * NH,
          wlT + (long)(l * ETY + e) * HH * HH,
          wrT + (long)(l * ETY + e) * HH * HH,
          bl + (long)(l * ETY + e) * HH,
          acc0 + DST[e] * NH);
    }
    relu_kernel<<<(int)(4 * NH / 256), 256, 0, stream>>>(acc0, x0, (int)(4 * NH));
  }

  // 4) output projection from x["reaction"] (type 2)
  out_gemm<<<NN / 16, 128, 0, stream>>>(x0 + 2 * NH, woT, bo, out);
}
